// EdgeConvNet_40535901339814
// MI455X (gfx1250) — compile-verified, matrix operand order checked by reading, not run on hardware
//
#include <hip/hip_runtime.h>
#include <math.h>

#define F_IN   32
#define EFEAT  6
#define H1C    128
#define H2C    64
#define GRAPHS 64
#define BN_EPS 1e-5f

typedef __attribute__((ext_vector_type(16))) __bf16 v16bf;
typedef __attribute__((ext_vector_type(8)))  float  v8f;

// ---------------------------------------------------------------------------
// A-fragment loader from LDS (wave32, 16x16x32 bf16).
// Lane 0-15 -> row M=lane&15, K in {kb..kb+7, kb+16..kb+23}; lanes 16-31 kb+=8.
// ---------------------------------------------------------------------------
__device__ __forceinline__ v16bf load_a_frag(const float* base, int ld, int row0, int kbase) {
    const int lane = threadIdx.x & 31;
    const float* p = base + (size_t)(row0 + (lane & 15)) * ld + kbase + ((lane >> 4) << 3);
    v16bf a;
#pragma unroll
    for (int j = 0; j < 8; ++j) a[j] = (__bf16)p[j];
#pragma unroll
    for (int j = 0; j < 8; ++j) a[8 + j] = (__bf16)p[16 + j];
    return a;
}

// ---------------------------------------------------------------------------
// Pack W3 [64, ldw] (f32) into per-lane B-fragment order (bf16):
// tile t (16 cols), K-half kh, lane l -> 16 contiguous bf16 (32B).
// One fragment load in the hot loop becomes a single 32B vector load.
// ---------------------------------------------------------------------------
__global__ __launch_bounds__(64)
void pack_w3(const float* __restrict__ W3, int ldw, __bf16* __restrict__ out) {
    const int lane = threadIdx.x & 31;
    const int kh = threadIdx.x >> 5;          // 0 or 1 (K half)
    const int t = blockIdx.x;                 // 16-column tile
    const int col = t * 16 + (lane & 15);
    const int kb = kh * 32 + ((lane >> 4) << 3);
    __bf16* dst = out + ((size_t)(t * 2 + kh) * 32 + lane) * 16;
#pragma unroll
    for (int j = 0; j < 8; ++j) dst[j] = (__bf16)W3[(size_t)(kb + j) * ldw + col];
#pragma unroll
    for (int j = 0; j < 8; ++j) dst[8 + j] = (__bf16)W3[(size_t)(kb + 16 + j) * ldw + col];
}

// ---------------------------------------------------------------------------
// Zero scratch accumulators
// ---------------------------------------------------------------------------
__global__ void zero_kernel(float* __restrict__ p, long n) {
    long i = (long)blockIdx.x * blockDim.x + threadIdx.x;
    if (i < n) p[i] = 0.0f;
}

// ---------------------------------------------------------------------------
// EdgeNet hidden: h2 = relu(relu(ea@W1+b1)@W2+b2)  [E,6] -> [E,64]
// ---------------------------------------------------------------------------
__global__ __launch_bounds__(128)
void edgenet_hidden(const float* __restrict__ ea,
                    const float* __restrict__ W1, const float* __restrict__ b1,
                    const float* __restrict__ W2, const float* __restrict__ b2,
                    float* __restrict__ h2out, int E) {
    __shared__ float sW1[EFEAT * 64];
    __shared__ float sb1[64];
    __shared__ float sW2[64 * 64];
    __shared__ float sb2[64];
    for (int t = threadIdx.x; t < EFEAT * 64; t += blockDim.x) sW1[t] = W1[t];
    for (int t = threadIdx.x; t < 64; t += blockDim.x) { sb1[t] = b1[t]; sb2[t] = b2[t]; }
    for (int t = threadIdx.x; t < 64 * 64; t += blockDim.x) sW2[t] = W2[t];
    __syncthreads();

    const int e = blockIdx.x * blockDim.x + threadIdx.x;
    if (e >= E) return;
    float a[EFEAT];
#pragma unroll
    for (int k = 0; k < EFEAT; ++k) a[k] = ea[(size_t)e * EFEAT + k];
    float h1r[64];
#pragma unroll 4
    for (int j = 0; j < 64; ++j) {
        float s = sb1[j];
#pragma unroll
        for (int k = 0; k < EFEAT; ++k) s += a[k] * sW1[k * 64 + j];
        h1r[j] = s > 0.0f ? s : 0.0f;
    }
    float* out = h2out + (size_t)e * 64;
    for (int j = 0; j < 64; ++j) {
        float s = sb2[j];
#pragma unroll
        for (int k = 0; k < 64; ++k) s += h1r[k] * sW2[k * 64 + j];
        out[j] = s > 0.0f ? s : 0.0f;
    }
}

// ---------------------------------------------------------------------------
// Fused: per-edge weight GEMM (relu(h2@W3+b3)) + per-edge matvec + scatter-add.
// One wave per 32 edges. K=64 via two v_wmma_f32_16x16x32_bf16 per 16x16 tile.
// B operands come pre-swizzled/pre-converted (Wp) -> one 32B load per fragment.
// ---------------------------------------------------------------------------
template <int IN_C, int OUT_C>
__global__ __launch_bounds__(32)
void fused_nnconv(const float* __restrict__ hsrc,  // [N, IN_C] node features
                  const float* __restrict__ h2e,   // [E, 64] edgenet hidden
                  const v16bf* __restrict__ Wp,    // packed W3 fragments
                  const float* __restrict__ b3,    // [IN_C*OUT_C]
                  const int* __restrict__ src, const int* __restrict__ dst,
                  float* __restrict__ agg,         // [N, OUT_C] pre-zeroed
                  int E) {
    __shared__ float sh_h2[32][65];        // +1 pad vs 64-bank LDS
    __shared__ float sh_x[32][IN_C + 1];
    __shared__ int   sh_dst[32];

    const int lane = threadIdx.x;
    const int e0 = blockIdx.x * 32;

    {   // stage this tile's edge hiddens + gathered source-node features
        const int e = e0 + lane;
        const bool valid = e < E;
        const int s = valid ? src[e] : 0;
        sh_dst[lane] = valid ? dst[e] : -1;
        const float4* hp = (const float4*)(h2e + (size_t)e * 64);
#pragma unroll
        for (int j = 0; j < 16; ++j) {
            float4 v = valid ? hp[j] : make_float4(0.f, 0.f, 0.f, 0.f);
            sh_h2[lane][j * 4 + 0] = v.x; sh_h2[lane][j * 4 + 1] = v.y;
            sh_h2[lane][j * 4 + 2] = v.z; sh_h2[lane][j * 4 + 3] = v.w;
        }
        const float4* xp = (const float4*)(hsrc + (size_t)s * IN_C);
#pragma unroll
        for (int j = 0; j < IN_C / 4; ++j) {
            float4 v = valid ? xp[j] : make_float4(0.f, 0.f, 0.f, 0.f);
            sh_x[lane][j * 4 + 0] = v.x; sh_x[lane][j * 4 + 1] = v.y;
            sh_x[lane][j * 4 + 2] = v.z; sh_x[lane][j * 4 + 3] = v.w;
        }
    }
    __syncthreads();

    // A fragments: 2 M-tiles x 2 K-halves, reused for every W3 column tile.
    v16bf a[2][2];
#pragma unroll
    for (int mt = 0; mt < 2; ++mt)
#pragma unroll
        for (int kh = 0; kh < 2; ++kh)
            a[mt][kh] = load_a_frag(&sh_h2[0][0], 65, mt * 16, kh * 32);

    constexpr int OT = OUT_C / 16;
    float msg[2][OT][8];
#pragma unroll
    for (int mt = 0; mt < 2; ++mt)
#pragma unroll
        for (int ot = 0; ot < OT; ++ot)
#pragma unroll
            for (int m = 0; m < 8; ++m) msg[mt][ot][m] = 0.0f;

    const int mrow = (lane >> 4) << 3;   // D-fragment row offset per lane half
    const int ncol = lane & 15;

    for (int i = 0; i < IN_C; ++i) {
        // prefetch next K-slice of packed B fragments (global_prefetch_b8)
        if (i + 1 < IN_C)
            __builtin_prefetch((const void*)(Wp + (size_t)(i + 1) * OT * 2 * 32), 0, 1);

        // per-row source-feature scalar x[src[e], i]
        float xv[2][8];
#pragma unroll
        for (int mt = 0; mt < 2; ++mt)
#pragma unroll
            for (int m = 0; m < 8; ++m)
                xv[mt][m] = sh_x[mt * 16 + mrow + m][i];

#pragma unroll
        for (int ot = 0; ot < OT; ++ot) {
            const int tile = i * OT + ot;
            v16bf b0 = Wp[((size_t)tile * 2 + 0) * 32 + lane];   // 32B vector load
            v16bf b1 = Wp[((size_t)tile * 2 + 1) * 32 + lane];
            const float bias = b3[tile * 16 + ncol];
#pragma unroll
            for (int mt = 0; mt < 2; ++mt) {
                v8f c = {0.f, 0.f, 0.f, 0.f, 0.f, 0.f, 0.f, 0.f};
                c = __builtin_amdgcn_wmma_f32_16x16x32_bf16(false, a[mt][0], false, b0,
                                                            (short)0, c, false, false);
                c = __builtin_amdgcn_wmma_f32_16x16x32_bf16(false, a[mt][1], false, b1,
                                                            (short)0, c, false, false);
#pragma unroll
                for (int m = 0; m < 8; ++m) {
                    float w = c[m] + bias;             // + b3
                    w = w > 0.0f ? w : 0.0f;           // relu -> per-edge weight
                    msg[mt][ot][m] += xv[mt][m] * w;   // einsum 'ei,eio->eo'
                }
            }
        }
    }

    // segment_sum(msg, dst): scatter with HW f32 atomics
#pragma unroll
    for (int mt = 0; mt < 2; ++mt)
#pragma unroll
        for (int m = 0; m < 8; ++m) {
            const int r = mt * 16 + mrow + m;
            const int d = sh_dst[r];
            if (d >= 0) {
#pragma unroll
                for (int ot = 0; ot < OT; ++ot)
                    unsafeAtomicAdd(&agg[(size_t)d * OUT_C + ot * 16 + ncol],
                                    msg[mt][ot][m]);
            }
        }
}

// ---------------------------------------------------------------------------
// Node update: h = bn(relu(agg + x@Wroot + bias)); optional pooled scatter.
// ---------------------------------------------------------------------------
template <int IN_C, int OUT_C, bool POOL>
__global__ __launch_bounds__(OUT_C)
void node_update(const float* __restrict__ xin, const float* __restrict__ agg,
                 const float* __restrict__ Wroot, const float* __restrict__ bias,
                 const float* __restrict__ gamma, const float* __restrict__ beta,
                 const float* __restrict__ mean, const float* __restrict__ var,
                 const int* __restrict__ batch, float* __restrict__ pooled,
                 float* __restrict__ hout, int N) {
    __shared__ float sx[IN_C];
    const int n = blockIdx.x;
    const int o = threadIdx.x;
    for (int t = o; t < IN_C; t += OUT_C) sx[t] = xin[(size_t)n * IN_C + t];
    __syncthreads();

    float s = agg[(size_t)n * OUT_C + o] + bias[o];
#pragma unroll 4
    for (int i = 0; i < IN_C; ++i) s += sx[i] * Wroot[i * OUT_C + o];
    s = s > 0.0f ? s : 0.0f;
    s = (s - mean[o]) * (gamma[o] * rsqrtf(var[o] + BN_EPS)) + beta[o];
    hout[(size_t)n * OUT_C + o] = s;
    if (POOL) unsafeAtomicAdd(&pooled[(size_t)batch[n] * OUT_C + o], s);
}

// ---------------------------------------------------------------------------
// Head: fc1+relu+bn3 -> fc2+relu -> fc3 -> log_softmax.  [G,64] -> [G,2]
// ---------------------------------------------------------------------------
__global__ __launch_bounds__(64)
void head_kernel(const float* __restrict__ pooled,
                 const float* __restrict__ fc1W, const float* __restrict__ fc1b,
                 const float* __restrict__ g3, const float* __restrict__ be3,
                 const float* __restrict__ mn3, const float* __restrict__ vr3,
                 const float* __restrict__ fc2W, const float* __restrict__ fc2b,
                 const float* __restrict__ fc3W, const float* __restrict__ fc3b,
                 float* __restrict__ out, int G) {
    const int t = threadIdx.x;
    if (t >= G) return;
    float r[64], h[64];
#pragma unroll
    for (int k = 0; k < 64; ++k) r[k] = pooled[t * 64 + k];
    for (int j = 0; j < 64; ++j) {
        float s = fc1b[j];
#pragma unroll
        for (int k = 0; k < 64; ++k) s += r[k] * fc1W[k * 64 + j];
        s = s > 0.0f ? s : 0.0f;
        h[j] = (s - mn3[j]) * (g3[j] * rsqrtf(vr3[j] + BN_EPS)) + be3[j];
    }
    for (int j = 0; j < 64; ++j) {
        float s = fc2b[j];
#pragma unroll
        for (int k = 0; k < 64; ++k) s += h[k] * fc2W[k * 64 + j];
        r[j] = s > 0.0f ? s : 0.0f;
    }
    float l0 = fc3b[0], l1 = fc3b[1];
#pragma unroll
    for (int k = 0; k < 64; ++k) { l0 += r[k] * fc3W[k * 2]; l1 += r[k] * fc3W[k * 2 + 1]; }
    const float mx = l0 > l1 ? l0 : l1;
    const float ls = logf(__expf(l0 - mx) + __expf(l1 - mx));
    out[t * 2 + 0] = (l0 - mx) - ls;
    out[t * 2 + 1] = (l1 - mx) - ls;
}

// ---------------------------------------------------------------------------
extern "C" void kernel_launch(void* const* d_in, const int* in_sizes, int n_in,
                              void* d_out, int out_size, void* d_ws, size_t ws_size,
                              hipStream_t stream) {
    const float* x  = (const float*)d_in[0];
    const int* eidx = (const int*)d_in[1];
    const float* ea = (const float*)d_in[2];
    const int* batch = (const int*)d_in[3];
    // params in dict insertion order
    const float* en1_W1 = (const float*)d_in[4];
    const float* en1_b1 = (const float*)d_in[5];
    const float* en1_W2 = (const float*)d_in[6];
    const float* en1_b2 = (const float*)d_in[7];
    const float* en1_W3 = (const float*)d_in[8];
    const float* en1_b3 = (const float*)d_in[9];
    const float* c1_root = (const float*)d_in[10];
    const float* c1_bias = (const float*)d_in[11];
    const float* bn1_g = (const float*)d_in[12];
    const float* bn1_b = (const float*)d_in[13];
    const float* bn1_m = (const float*)d_in[14];
    const float* bn1_v = (const float*)d_in[15];
    const float* en2_W1 = (const float*)d_in[16];
    const float* en2_b1 = (const float*)d_in[17];
    const float* en2_W2 = (const float*)d_in[18];
    const float* en2_b2 = (const float*)d_in[19];
    const float* en2_W3 = (const float*)d_in[20];
    const float* en2_b3 = (const float*)d_in[21];
    const float* c2_root = (const float*)d_in[22];
    const float* c2_bias = (const float*)d_in[23];
    const float* bn2_g = (const float*)d_in[24];
    const float* bn2_b = (const float*)d_in[25];
    const float* bn2_m = (const float*)d_in[26];
    const float* bn2_v = (const float*)d_in[27];
    const float* fc1_W = (const float*)d_in[28];
    const float* fc1_b = (const float*)d_in[29];
    const float* bn3_g = (const float*)d_in[30];
    const float* bn3_b = (const float*)d_in[31];
    const float* bn3_m = (const float*)d_in[32];
    const float* bn3_v = (const float*)d_in[33];
    const float* fc2_W = (const float*)d_in[34];
    const float* fc2_b = (const float*)d_in[35];
    const float* fc3_W = (const float*)d_in[36];
    const float* fc3_b = (const float*)d_in[37];

    const int N = in_sizes[0] / F_IN;
    const int E = in_sizes[1] / 2;
    const int G = out_size / 2;
    const int* src = eidx;
    const int* dst = eidx + E;

    // workspace layout (floats; every block is a multiple of 8 floats = 32B,
    // so the packed-bf16 v16bf regions stay 32B-aligned)
    float* ws = (float*)d_ws;
    float* agg1   = ws;                             // N*128
    float* agg2   = agg1 + (size_t)N * H1C;         // N*64
    float* pooled = agg2 + (size_t)N * H2C;         // G*64
    float* h2e1   = pooled + (size_t)GRAPHS * H2C;  // E*64
    float* h2e2   = h2e1 + (size_t)E * 64;          // E*64
    float* hn1    = h2e2 + (size_t)E * 64;          // N*128
    float* hn2    = hn1 + (size_t)N * H1C;          // N*64
    __bf16* wp1   = (__bf16*)(hn2 + (size_t)N * H2C);      // 64*F_IN*H1C bf16
    __bf16* wp2   = wp1 + (size_t)64 * F_IN * H1C;         // 64*H1C*H2C bf16
    (void)ws_size;

    const long nzero = (long)N * H1C + (long)N * H2C + (long)GRAPHS * H2C;
    zero_kernel<<<(int)((nzero + 255) / 256), 256, 0, stream>>>(agg1, nzero);

    // pre-swizzle W3 weights into bf16 WMMA B-fragment layout
    pack_w3<<<(F_IN * H1C) / 16, 64, 0, stream>>>(en1_W3, F_IN * H1C, wp1);
    pack_w3<<<(H1C * H2C) / 16, 64, 0, stream>>>(en2_W3, H1C * H2C, wp2);

    const int ebl = (E + 127) / 128;
    edgenet_hidden<<<ebl, 128, 0, stream>>>(ea, en1_W1, en1_b1, en1_W2, en1_b2, h2e1, E);
    edgenet_hidden<<<ebl, 128, 0, stream>>>(ea, en2_W1, en2_b1, en2_W2, en2_b2, h2e2, E);

    fused_nnconv<F_IN, H1C><<<(E + 31) / 32, 32, 0, stream>>>(
        x, h2e1, (const v16bf*)wp1, en1_b3, src, dst, agg1, E);

    node_update<F_IN, H1C, false><<<N, H1C, 0, stream>>>(
        x, agg1, c1_root, c1_bias, bn1_g, bn1_b, bn1_m, bn1_v,
        nullptr, nullptr, hn1, N);

    fused_nnconv<H1C, H2C><<<(E + 31) / 32, 32, 0, stream>>>(
        hn1, h2e2, (const v16bf*)wp2, en2_b3, src, dst, agg2, E);

    node_update<H1C, H2C, true><<<N, H2C, 0, stream>>>(
        hn1, agg2, c2_root, c2_bias, bn2_g, bn2_b, bn2_m, bn2_v,
        batch, pooled, hn2, N);

    head_kernel<<<1, 64, 0, stream>>>(pooled, fc1_W, fc1_b, bn3_g, bn3_b, bn3_m, bn3_v,
                                      fc2_W, fc2_b, fc3_W, fc3_b, (float*)d_out, G);
}
